// ClusterMemory_84447646974572
// MI455X (gfx1250) — compile-verified, matrix operand order checked by reading, not run on hardware
//
#include <hip/hip_runtime.h>
#include <hip/hip_bf16.h>

// Problem constants (from reference)
#define PP 16
#define CC 4
#define KK 8
#define NN 65536
#define DD 256
#define GG (PP*CC)           // 64 output rows
#define SCALE 2.5f           // 1/(K*TEMP) = 1/(8*0.05)
#define NWG 512              // GEMM workgroups (NN / 128)

typedef float    v8f  __attribute__((ext_vector_type(8)));
typedef _Float16 v16h __attribute__((ext_vector_type(16)));
typedef _Float16 v8h  __attribute__((ext_vector_type(8)));

// ---------------- workspace layout (floats) ----------------
// XBAR : [64][256]                      @ 0
// PM   : [64][512]  per-(row,wg) max    @ 16384
// PS   : [64][512]  per-(row,wg) sumexp @ 49152
// PMN  : [64][512]  per-(row,wg) min    @ 81920
// OWN  : [64]                           @ 114688
// ROWM : [64]                           @ 114752
// ROWS : [64]                           @ 114816
// ROWMN: [64]                           @ 114880
#define WS_XBAR  0
#define WS_PM    16384
#define WS_PS    49152
#define WS_PMN   81920
#define WS_OWN   114688
#define WS_ROWM  114752
#define WS_ROWS  114816
#define WS_ROWMN 114880

__device__ __forceinline__ void lse_merge(float& m, float& s, float om, float os) {
    float nm = fmaxf(m, om);
    s = s * __expf(m - nm) + os * __expf(om - nm);
    m = nm;
}

// ---------------------------------------------------------------------------
// Kernel 0: xbar[g][d] = SCALE * sum_k inputs[g*8+k][d] / ||inputs[g*8+k]||
// 64 blocks x 256 threads, one block per (p,c) group of 8 rows.
// ---------------------------------------------------------------------------
__global__ __launch_bounds__(256) void xbar_kernel(const float* __restrict__ in,
                                                   float* __restrict__ ws) {
    __shared__ float ldsW[KK][8];   // [k][wave] partial sums of squares
    __shared__ float ldsN[KK];      // inverse norms
    const int g    = blockIdx.x;
    const int tid  = threadIdx.x;
    const int lane = tid & 31;
    const int wave = tid >> 5;

    float vals[KK];
    for (int k = 0; k < KK; ++k) {
        float v = in[(size_t)(g * KK + k) * DD + tid];
        vals[k] = v;
        float p = v * v;
        #pragma unroll
        for (int m = 16; m >= 1; m >>= 1) p += __shfl_xor(p, m, 32);
        if (lane == 0) ldsW[k][wave] = p;
    }
    __syncthreads();
    if (tid < KK) {
        float s = 0.f;
        #pragma unroll
        for (int w = 0; w < 8; ++w) s += ldsW[tid][w];
        ldsN[tid] = rsqrtf(s);
    }
    __syncthreads();
    float acc = 0.f;
    #pragma unroll
    for (int k = 0; k < KK; ++k) acc += vals[k] * ldsN[k];
    ws[WS_XBAR + g * DD + tid] = acc * SCALE;
}

// ---------------------------------------------------------------------------
// Kernel 1: own[g] = dot(xbar[g], features[g>>2])   (fp32, 64 dots of 256)
// ---------------------------------------------------------------------------
__global__ __launch_bounds__(64) void own_kernel(const float* __restrict__ feats,
                                                 float* __restrict__ ws) {
    const int i = threadIdx.x;           // g = p*4 + c
    const int p = i >> 2;
    const float* xb = ws + WS_XBAR + i * DD;
    const float* fr = feats + (size_t)p * DD;
    float acc = 0.f;
    for (int d = 0; d < DD; ++d) acc += xb[d] * fr[d];
    ws[WS_OWN + i] = acc;
}

// ---------------------------------------------------------------------------
// Kernel 2: WMMA GEMM  proxies = xbar(64x256,f16) @ features.T  with fused
// per-row (max, sumexp, min) partials per workgroup.
// 512 blocks x 256 threads (8 waves). Each wave: 16 N-columns x all 64 rows.
// ---------------------------------------------------------------------------
__global__ __launch_bounds__(256) void gemm_kernel(const float* __restrict__ feats,
                                                   float* __restrict__ ws) {
    __shared__ _Float16 ldsA[GG * 264];         // 64 rows x 256 halves, pad 8
    __shared__ float redM[GG][8];
    __shared__ float redS[GG][8];
    __shared__ float redN[GG][8];

    const int tid  = threadIdx.x;
    const int lane = tid & 31;
    const int wave = tid >> 5;
    const int bid  = blockIdx.x;

    // stage xbar -> LDS as f16
    for (int i = tid; i < GG * DD; i += 256) {
        ldsA[(i >> 8) * 264 + (i & 255)] = (_Float16)ws[WS_XBAR + i];
    }
    __syncthreads();

    const int nCol = bid * 128 + wave * 16 + (lane & 15);
    const float* __restrict__ frow = feats + (size_t)nCol * DD;
    const int kOff = (lane & 16) ? 8 : 0;   // lanes 16-31 take K+8 / K+24

    v8f acc[4];
    #pragma unroll
    for (int t = 0; t < 4; ++t) acc[t] = (v8f)0.f;

    #pragma unroll
    for (int ks = 0; ks < 8; ++ks) {
        const int kb = ks * 32 + kOff;
        float4 b0 = *(const float4*)(frow + kb);
        float4 b1 = *(const float4*)(frow + kb + 4);
        float4 b2 = *(const float4*)(frow + kb + 16);
        float4 b3 = *(const float4*)(frow + kb + 20);
        if (ks < 7) __builtin_prefetch((const void*)(frow + kb + 32), 0, 0);

        v16h bf;
        bf[0]=(_Float16)b0.x;  bf[1]=(_Float16)b0.y;  bf[2]=(_Float16)b0.z;  bf[3]=(_Float16)b0.w;
        bf[4]=(_Float16)b1.x;  bf[5]=(_Float16)b1.y;  bf[6]=(_Float16)b1.z;  bf[7]=(_Float16)b1.w;
        bf[8]=(_Float16)b2.x;  bf[9]=(_Float16)b2.y;  bf[10]=(_Float16)b2.z; bf[11]=(_Float16)b2.w;
        bf[12]=(_Float16)b3.x; bf[13]=(_Float16)b3.y; bf[14]=(_Float16)b3.z; bf[15]=(_Float16)b3.w;

        #pragma unroll
        for (int t = 0; t < 4; ++t) {
            const _Float16* ar = ldsA + (t * 16 + (lane & 15)) * 264 + kb;
            v8h a0 = *(const v8h*)(ar);
            v8h a1 = *(const v8h*)(ar + 16);
            v16h af;
            #pragma unroll
            for (int j = 0; j < 8; ++j) { af[j] = a0[j]; af[8 + j] = a1[j]; }
            acc[t] = __builtin_amdgcn_wmma_f32_16x16x32_f16(
                false, af, false, bf, (short)0, acc[t], false, false);
        }
    }

    // fused per-row (max, sumexp, min) over this wave's 16 columns
    #pragma unroll
    for (int t = 0; t < 4; ++t) {
        #pragma unroll
        for (int j = 0; j < 8; ++j) {
            float v = acc[t][j];
            float m = v, s = 1.0f, mn = v;
            #pragma unroll
            for (int msk = 1; msk <= 8; msk <<= 1) {   // stays within 16-lane halves
                float om  = __shfl_xor(m,  msk, 32);
                float os  = __shfl_xor(s,  msk, 32);
                float omn = __shfl_xor(mn, msk, 32);
                lse_merge(m, s, om, os);
                mn = fminf(mn, omn);
            }
            if ((lane & 15) == 0) {
                int row = t * 16 + j + ((lane >> 4) << 3);
                redM[row][wave] = m; redS[row][wave] = s; redN[row][wave] = mn;
            }
        }
    }
    __syncthreads();

    if (tid < GG) {
        float m = redM[tid][0], s = redS[tid][0], mn = redN[tid][0];
        #pragma unroll
        for (int w = 1; w < 8; ++w) {
            lse_merge(m, s, redM[tid][w], redS[tid][w]);
            mn = fminf(mn, redN[tid][w]);
        }
        ws[WS_PM  + tid * NWG + bid] = m;
        ws[WS_PS  + tid * NWG + bid] = s;
        ws[WS_PMN + tid * NWG + bid] = mn;
    }
}

// ---------------------------------------------------------------------------
// Kernel 3: reduce 512 partials per row -> (rowM, rowS, rowMN). 64 blocks.
// ---------------------------------------------------------------------------
__global__ __launch_bounds__(256) void reduce_kernel(float* __restrict__ ws) {
    __shared__ float ldsM[8], ldsS[8], ldsN[8];
    const int r    = blockIdx.x;
    const int tid  = threadIdx.x;
    const int lane = tid & 31;
    const int wave = tid >> 5;

    float m  = ws[WS_PM  + r * NWG + tid];
    float s  = ws[WS_PS  + r * NWG + tid];
    float mn = ws[WS_PMN + r * NWG + tid];
    float m2  = ws[WS_PM  + r * NWG + tid + 256];
    float s2  = ws[WS_PS  + r * NWG + tid + 256];
    float mn2 = ws[WS_PMN + r * NWG + tid + 256];
    lse_merge(m, s, m2, s2);
    mn = fminf(mn, mn2);

    #pragma unroll
    for (int msk = 1; msk <= 16; msk <<= 1) {
        float om  = __shfl_xor(m,  msk, 32);
        float os  = __shfl_xor(s,  msk, 32);
        float omn = __shfl_xor(mn, msk, 32);
        lse_merge(m, s, om, os);
        mn = fminf(mn, omn);
    }
    if (lane == 0) { ldsM[wave] = m; ldsS[wave] = s; ldsN[wave] = mn; }
    __syncthreads();
    if (tid == 0) {
        float M = ldsM[0], S = ldsS[0], MN = ldsN[0];
        #pragma unroll
        for (int w = 1; w < 8; ++w) {
            lse_merge(M, S, ldsM[w], ldsS[w]);
            MN = fminf(MN, ldsN[w]);
        }
        ws[WS_ROWM  + r] = M;
        ws[WS_ROWS  + r] = S;
        ws[WS_ROWMN + r] = MN;
    }
}

// ---------------------------------------------------------------------------
// Kernel 4: final loss. loss[p] = log( e^{pos-M} + sum_c(S_c e^{m_c-M} - e^{mn_c-M}) ) + M - pos
// one wave; lanes 0..15 = p. Output: mean over p.
// ---------------------------------------------------------------------------
__global__ __launch_bounds__(32) void final_kernel(const float* __restrict__ ws,
                                                   float* __restrict__ out) {
    const int p = threadIdx.x;
    float loss = 0.f;
    if (p < PP) {
        // argmin over c of own[p][c]; pos = that own value
        float pos = ws[WS_OWN + p * CC];
        #pragma unroll
        for (int c = 1; c < CC; ++c) {
            float o = ws[WS_OWN + p * CC + c];
            if (o < pos) pos = o;
        }
        float M = pos;
        #pragma unroll
        for (int c = 0; c < CC; ++c) M = fmaxf(M, ws[WS_ROWM + p * CC + c]);
        float total = __expf(pos - M);
        #pragma unroll
        for (int c = 0; c < CC; ++c) {
            int r = p * CC + c;
            total += ws[WS_ROWS + r] * __expf(ws[WS_ROWM + r] - M);
            total -= __expf(ws[WS_ROWMN + r] - M);
        }
        loss = logf(total) + M - pos;
    }
    #pragma unroll
    for (int msk = 16; msk >= 1; msk >>= 1) loss += __shfl_xor(loss, msk, 32);
    if (p == 0) out[0] = loss * (1.0f / PP);
}

// ---------------------------------------------------------------------------
extern "C" void kernel_launch(void* const* d_in, const int* in_sizes, int n_in,
                              void* d_out, int out_size, void* d_ws, size_t ws_size,
                              hipStream_t stream) {
    const float* inputs = (const float*)d_in[0];   // (512, 256) f32
    const float* feats  = (const float*)d_in[4];   // (65536, 256) f32, pre-normalized
    float* ws  = (float*)d_ws;
    float* out = (float*)d_out;

    xbar_kernel  <<<GG,  256, 0, stream>>>(inputs, ws);
    own_kernel   <<<1,    64, 0, stream>>>(feats, ws);
    gemm_kernel  <<<NWG, 256, 0, stream>>>(feats, ws);
    reduce_kernel<<<GG,  256, 0, stream>>>(ws);
    final_kernel <<<1,    32, 0, stream>>>(ws, out);
    (void)in_sizes; (void)n_in; (void)out_size; (void)ws_size;
}
